// RoPEMultiHeadAttention_66718021976475
// MI455X (gfx1250) — matrix operand order repaired and checked by reading, not compiled, not run on hardware
//
#include <hip/hip_runtime.h>
#include <hip/hip_bf16.h>

// ---------------------------------------------------------------------------
// RoPE "head-mixing" attention block for MI455X (gfx1250, wave32, WMMA).
// Dominant cost: four 8192x2304x2304 GEMMs -> bf16 WMMA w/ f32 accumulate,
// fed by CDNA5 async global->LDS copies (ASYNCcnt) with double buffering.
// ---------------------------------------------------------------------------

#define BATCH   4
#define SEQ     2048
#define HID     2304
#define HEADS   16
#define DH      (HID / HEADS)          // 144
#define TOKENS  (BATCH * SEQ)          // 8192
#define KPAIRS  (HID / 2)              // 1152 uint (bf16x2) pairs per row

#define USE_ASYNC_LDS 1                // CDNA5 global_load_async_to_lds path

typedef __attribute__((ext_vector_type(16))) __bf16 v16bf;
typedef __attribute__((ext_vector_type(8)))  float  v8f;

union BFrag { unsigned int u[8]; v16bf v; };

__device__ __forceinline__ unsigned short f32_to_bf16(float f) {
    union { float f; unsigned int u; } c; c.f = f;
    unsigned int r = c.u + 0x7FFFu + ((c.u >> 16) & 1u);   // round-to-nearest-even
    return (unsigned short)(r >> 16);
}

#if USE_ASYNC_LDS
// LDS byte offset of a generic pointer into a __shared__ array (AS3 ptrs are
// 32-bit LDS byte offsets on amdgcn).
__device__ __forceinline__ unsigned lds_byte_off(const void* p) {
    return (unsigned)(unsigned long long)
        (__attribute__((address_space(3))) const void*)p;
}
// Per-lane 16B global -> LDS async copy (tracked by ASYNCcnt).
__device__ __forceinline__ void async_copy_b128(unsigned ldsOff,
                                                const unsigned int* g) {
    asm volatile("global_load_async_to_lds_b128 %0, %1, off"
                 :: "v"(ldsOff), "v"(g) : "memory");
}
__device__ __forceinline__ void wait_async_zero() {
    asm volatile("s_wait_asynccnt 0x0" ::: "memory");
}
#endif

// ---------------------------------------------------------------------------
// f32 -> bf16 bulk convert
// ---------------------------------------------------------------------------
__global__ void cvt_bf16_kernel(const float* __restrict__ src,
                                unsigned short* __restrict__ dst, int n) {
    int i = blockIdx.x * blockDim.x + threadIdx.x;
    if (i < n) dst[i] = f32_to_bf16(src[i]);
}

// ---------------------------------------------------------------------------
// GEMM: C[m][n] = sum_k A[m][k] * W[n][k] + bias[n]
// A: [M][HID] bf16 (uint pairs), W: [HID][HID] bf16 (uint pairs).
// Block tile 128x256, BK=32; 8 waves as (wm 0..1, wn 0..3), each wave owns a
// 64x64 sub-tile = 4x4 v_wmma_f32_16x16x32_bf16 accumulators (16 WMMA/step).
// Double-buffered LDS filled with global_load_async_to_lds_b128.
// ---------------------------------------------------------------------------
#define BM 128
#define BN 256
#define BK 32
#define NK   (HID / BK)   // 72 k-steps
#define KPT  (BK / 2)     // 16 uint pairs per k-tile row
#define LPIT 20           // LDS row pitch in uints (80B: 16B-aligned, conflict-free)
#define ASZ  (BM * LPIT)  // one A buffer, uints
#define BSZ  (BN * LPIT)  // one B buffer, uints

__global__ __launch_bounds__(256)
void gemm_bf16_wmma_kernel(const unsigned int* __restrict__ A,
                           const unsigned int* __restrict__ W,
                           const float* __restrict__ bias,
                           float* __restrict__ C) {
    __shared__ __align__(16) unsigned int As[2 * ASZ];   // 20 KB
    __shared__ __align__(16) unsigned int Bs[2 * BSZ];   // 40 KB

    const int tid  = threadIdx.x;
    const int lane = tid & 31;
    const int wid  = tid >> 5;
    const int wm   = wid & 1;     // wave row (0..1) -> 64 rows each
    const int wn   = wid >> 1;    // wave col (0..3) -> 64 cols each
    const int hl   = lane & 15;
    const int hi   = lane >> 4;

    const int n0 = blockIdx.x * BN;
    const int m0 = blockIdx.y * BM;

    // global->LDS chunk mapping (uint4 chunks of 4 pairs):
    //   A tile: 128 rows x 4 chunks = 512 -> 2 per thread
    //   B tile: 256 rows x 4 chunks = 1024 -> 4 per thread
    const int rA0 = tid >> 2,          qA0 = tid & 3;
    const int rA1 = (tid + 256) >> 2,  qA1 = (tid + 256) & 3;
    int rB[4], qB[4];
#pragma unroll
    for (int i = 0; i < 4; ++i) {
        const int c = tid + 256 * i;
        rB[i] = c >> 2; qB[i] = c & 3;
    }
    const size_t gA0 = (size_t)rA0 * KPAIRS + qA0 * 4;
    const size_t gA1 = (size_t)rA1 * KPAIRS + qA1 * 4;
    size_t gBo[4];
#pragma unroll
    for (int i = 0; i < 4; ++i) gBo[i] = (size_t)rB[i] * KPAIRS + qB[i] * 4;

#if USE_ASYNC_LDS
    const unsigned aL0 = lds_byte_off(&As[rA0 * LPIT + qA0 * 4]);
    const unsigned aL1 = lds_byte_off(&As[rA1 * LPIT + qA1 * 4]);
    unsigned bL[4];
#pragma unroll
    for (int i = 0; i < 4; ++i)
        bL[i] = lds_byte_off(&Bs[rB[i] * LPIT + qB[i] * 4]);
#endif

    v8f acc[4][4];
    const v8f vzero = {0.f,0.f,0.f,0.f,0.f,0.f,0.f,0.f};
#pragma unroll
    for (int i = 0; i < 4; ++i)
#pragma unroll
        for (int j = 0; j < 4; ++j) acc[i][j] = vzero;

    const unsigned int* gAbase = A + (size_t)m0 * KPAIRS;
    const unsigned int* gBbase = W + (size_t)n0 * KPAIRS;

#if USE_ASYNC_LDS
    // prologue: fill buffer 0 with tile 0
    async_copy_b128(aL0, gAbase + gA0);
    async_copy_b128(aL1, gAbase + gA1);
#pragma unroll
    for (int i = 0; i < 4; ++i) async_copy_b128(bL[i], gBbase + gBo[i]);
#endif

    unsigned buf = 0;
    for (int kt = 0; kt < NK; ++kt) {
#if USE_ASYNC_LDS
        wait_async_zero();       // this wave's fills for tile kt have landed
        __syncthreads();         // => every wave's fills landed; alt buf free
        if (kt + 1 < NK) {       // overlap next tile's fill with compute
            const unsigned int* nA = gAbase + (size_t)(kt + 1) * KPT;
            const unsigned int* nB = gBbase + (size_t)(kt + 1) * KPT;
            const unsigned bo = (buf ^ 1) * (ASZ * 4);
            const unsigned bo2 = (buf ^ 1) * (BSZ * 4);
            async_copy_b128(aL0 + bo, nA + gA0);
            async_copy_b128(aL1 + bo, nA + gA1);
#pragma unroll
            for (int i = 0; i < 4; ++i) async_copy_b128(bL[i] + bo2, nB + gBo[i]);
        }
#else
        {
            const unsigned int* gA = gAbase + (size_t)kt * KPT;
            const unsigned int* gB = gBbase + (size_t)kt * KPT;
            uint4 a0 = *(const uint4*)(gA + gA0);
            uint4 a1 = *(const uint4*)(gA + gA1);
            uint4 b0 = *(const uint4*)(gB + gBo[0]);
            uint4 b1 = *(const uint4*)(gB + gBo[1]);
            uint4 b2 = *(const uint4*)(gB + gBo[2]);
            uint4 b3 = *(const uint4*)(gB + gBo[3]);
            __syncthreads();
            *(uint4*)&As[buf * ASZ + rA0 * LPIT + qA0 * 4] = a0;
            *(uint4*)&As[buf * ASZ + rA1 * LPIT + qA1 * 4] = a1;
            *(uint4*)&Bs[buf * BSZ + rB[0] * LPIT + qB[0] * 4] = b0;
            *(uint4*)&Bs[buf * BSZ + rB[1] * LPIT + qB[1] * 4] = b1;
            *(uint4*)&Bs[buf * BSZ + rB[2] * LPIT + qB[2] * 4] = b2;
            *(uint4*)&Bs[buf * BSZ + rB[3] * LPIT + qB[3] * 4] = b3;
            __syncthreads();
        }
#endif
        const unsigned int* Asb = As + buf * ASZ;
        const unsigned int* Bsb = Bs + buf * BSZ;

        // A fragments: 16x32 bf16 ISA layout: lane row M=hl; VGPR j<4 holds
        // K-pair j+4*hi, VGPR j>=4 holds pair 8+(j-4)+4*hi.
        BFrag afr[4], bfr[4];
#pragma unroll
        for (int tm = 0; tm < 4; ++tm) {
            const int row = wm * 64 + tm * 16 + hl;
#pragma unroll
            for (int j = 0; j < 8; ++j) {
                const int p = (j < 4) ? (j + 4 * hi) : (4 + j + 4 * hi);
                afr[tm].u[j] = Asb[row * LPIT + p];
            }
        }
        // B fragments: 32x16 bf16: lane hl = N column; VGPR j<4 = K-pair j+4*hi, j>=4 = pair 8+(j-4)+4*hi.
#pragma unroll
        for (int tn = 0; tn < 4; ++tn) {
            const int col = wn * 64 + tn * 16 + hl;
#pragma unroll
            for (int j = 0; j < 8; ++j)
                bfr[tn].u[j] = Bsb[col * LPIT + ((j < 4) ? (j + 4 * hi) : (4 + j + 4 * hi))];
        }

#pragma unroll
        for (int tm = 0; tm < 4; ++tm)
#pragma unroll
            for (int tn = 0; tn < 4; ++tn)
                acc[tm][tn] = __builtin_amdgcn_wmma_f32_16x16x32_bf16(
                    false, afr[tm].v, false, bfr[tn].v,
                    (short)0, acc[tm][tn], false, false);

        buf ^= 1;
    }

    // Epilogue: C/D layout -> element r at lane: M = r + 8*hi, N = hl
#pragma unroll
    for (int tm = 0; tm < 4; ++tm) {
#pragma unroll
        for (int tn = 0; tn < 4; ++tn) {
            const int col = n0 + wn * 64 + tn * 16 + hl;
            const float bv = bias[col];
#pragma unroll
            for (int r = 0; r < 8; ++r) {
                const int row = m0 + wm * 64 + tm * 16 + r + 8 * hi;
                C[(size_t)row * HID + col] = acc[tm][tn][r] + bv;
            }
        }
    }
}

// ---------------------------------------------------------------------------
// RoPE over full hidden dim (d/2 pairs), in-place on f32 q and k.
// ---------------------------------------------------------------------------
__global__ __launch_bounds__(256)
void rope_kernel(float* __restrict__ q, float* __restrict__ k) {
    const int token = blockIdx.x;
    const float pos = (float)(token % SEQ);
    const size_t base = (size_t)token * HID;
    for (int j = threadIdx.x; j < HID / 2; j += 256) {
        // omega = theta^{-(2j/d)}
        const float omega = __expf(-9.210340371976184f * (2.0f * (float)j) / (float)HID);
        float s, c;
        __sincosf(pos * omega, &s, &c);
        const size_t i = base + 2 * j;
        float q0 = q[i], q1 = q[i + 1];
        float k0 = k[i], k1 = k[i + 1];
        q[i]     = c * q0 - s * q1;
        q[i + 1] = s * q0 + c * q1;
        k[i]     = c * k0 - s * k1;
        k[i + 1] = s * k0 + c * k1;
    }
}

// ---------------------------------------------------------------------------
// Head-mixing attention, one block per token:
//   logits[n][m] = (Q_t[n,:].K_t[m,:]) / sqrt(Dh); softmax over m;
//   out[n][d] = sum_m w[n][m] * V_t[m][d].   16x16 logits -> VALU is plenty.
// Emits bf16 for the output-projection GEMM.
// ---------------------------------------------------------------------------
__global__ __launch_bounds__(256)
void attn_kernel(const float* __restrict__ q, const float* __restrict__ k,
                 const float* __restrict__ v, unsigned short* __restrict__ out) {
    __shared__ float Qs[HID], Ks[HID], Vs[HID];
    __shared__ float Lg[HEADS * HEADS], Wt[HEADS * HEADS];
    const int tid = threadIdx.x;
    const size_t base = (size_t)blockIdx.x * HID;

    for (int i = tid; i < HID; i += 256) {
        Qs[i] = q[base + i];
        Ks[i] = k[base + i];
        Vs[i] = v[base + i];
    }
    __syncthreads();

    const int n = tid >> 4, m = tid & 15;
    float a = 0.f;
#pragma unroll 8
    for (int d = 0; d < DH; ++d) a += Qs[n * DH + d] * Ks[m * DH + d];
    Lg[tid] = a * (1.0f / 12.0f);            // 1/sqrt(144)
    __syncthreads();

    float mx = -3.402823466e38f;
#pragma unroll
    for (int j = 0; j < HEADS; ++j) mx = fmaxf(mx, Lg[n * HEADS + j]);
    const float e = __expf(Lg[tid] - mx);
    Wt[tid] = e;
    __syncthreads();
    float sum = 0.f;
#pragma unroll
    for (int j = 0; j < HEADS; ++j) sum += Wt[n * HEADS + j];
    Lg[tid] = e / sum;                       // normalized weights
    __syncthreads();

    for (int i = tid; i < HID; i += 256) {
        const int nn = i / DH, dd = i % DH;
        float o = 0.f;
#pragma unroll
        for (int mm = 0; mm < HEADS; ++mm)
            o += Lg[nn * HEADS + mm] * Vs[mm * DH + dd];
        out[base + i] = f32_to_bf16(o);
    }
}

// ---------------------------------------------------------------------------
// y = x + proj ; LayerNorm(y) * g + b -> d_out.  One block per token.
// ---------------------------------------------------------------------------
__global__ __launch_bounds__(256)
void ln_kernel(const float* __restrict__ x, const float* __restrict__ proj,
               const float* __restrict__ g, const float* __restrict__ b,
               float* __restrict__ out) {
    __shared__ float Ys[HID];
    __shared__ float r1[256], r2[256];
    const int tid = threadIdx.x;
    const size_t base = (size_t)blockIdx.x * HID;

    float s1 = 0.f, s2 = 0.f;
    for (int i = tid; i < HID; i += 256) {
        const float yv = x[base + i] + proj[base + i];
        Ys[i] = yv;
        s1 += yv;
        s2 += yv * yv;
    }
    r1[tid] = s1; r2[tid] = s2;
    __syncthreads();
    for (int o = 128; o > 0; o >>= 1) {
        if (tid < o) { r1[tid] += r1[tid + o]; r2[tid] += r2[tid + o]; }
        __syncthreads();
    }
    const float mean = r1[0] * (1.0f / HID);
    const float var  = r2[0] * (1.0f / HID) - mean * mean;
    const float rstd = rsqrtf(var + 1e-5f);
    for (int i = tid; i < HID; i += 256)
        out[base + i] = (Ys[i] - mean) * rstd * g[i] + b[i];
}

// ---------------------------------------------------------------------------
// Launch
// ---------------------------------------------------------------------------
extern "C" void kernel_launch(void* const* d_in, const int* in_sizes, int n_in,
                              void* d_out, int out_size, void* d_ws, size_t ws_size,
                              hipStream_t stream) {
    const float* x    = (const float*)d_in[0];
    const float* wq   = (const float*)d_in[1];
    const float* bq   = (const float*)d_in[2];
    const float* wk   = (const float*)d_in[3];
    const float* bk   = (const float*)d_in[4];
    const float* wv   = (const float*)d_in[5];
    const float* bv   = (const float*)d_in[6];
    const float* wo   = (const float*)d_in[7];
    const float* bo   = (const float*)d_in[8];
    const float* ln_g = (const float*)d_in[9];
    const float* ln_b = (const float*)d_in[10];
    float* out = (float*)d_out;

    // Workspace layout (all offsets multiples of 16B):
    const size_t WB  = (size_t)HID * HID * 2;      // one bf16 weight matrix
    const size_t XBS = (size_t)TOKENS * HID * 2;   // bf16 activations
    const size_t QF  = (size_t)TOKENS * HID * 4;   // f32 activations
    char* ws = (char*)d_ws;
    unsigned short* wq_b = (unsigned short*)(ws);
    unsigned short* wk_b = (unsigned short*)(ws + WB);
    unsigned short* wv_b = (unsigned short*)(ws + 2 * WB);
    unsigned short* wo_b = (unsigned short*)(ws + 3 * WB);
    unsigned short* x_b  = (unsigned short*)(ws + 4 * WB);
    float*          qf   = (float*)(ws + 4 * WB + XBS);
    float*          kf   = (float*)(ws + 4 * WB + XBS + QF);
    float*          vf   = (float*)(ws + 4 * WB + XBS + 2 * QF);
    unsigned short* at_b = (unsigned short*)(ws + 4 * WB + XBS + 3 * QF);
    float*          proj = qf;   // q is dead after attn_kernel -> reuse

    const int NX = TOKENS * HID;       // 18,874,368
    const int NW = HID * HID;          // 5,308,416

    // 1) bf16 conversions
    cvt_bf16_kernel<<<(NX + 255) / 256, 256, 0, stream>>>(x,  x_b,  NX);
    cvt_bf16_kernel<<<(NW + 255) / 256, 256, 0, stream>>>(wq, wq_b, NW);
    cvt_bf16_kernel<<<(NW + 255) / 256, 256, 0, stream>>>(wk, wk_b, NW);
    cvt_bf16_kernel<<<(NW + 255) / 256, 256, 0, stream>>>(wv, wv_b, NW);
    cvt_bf16_kernel<<<(NW + 255) / 256, 256, 0, stream>>>(wo, wo_b, NW);

    // 2) Q/K/V projections (WMMA)
    dim3 ggrid(HID / BN, TOKENS / BM);     // (9, 64)
    gemm_bf16_wmma_kernel<<<ggrid, 256, 0, stream>>>(
        (const unsigned int*)x_b, (const unsigned int*)wq_b, bq, qf);
    gemm_bf16_wmma_kernel<<<ggrid, 256, 0, stream>>>(
        (const unsigned int*)x_b, (const unsigned int*)wk_b, bk, kf);
    gemm_bf16_wmma_kernel<<<ggrid, 256, 0, stream>>>(
        (const unsigned int*)x_b, (const unsigned int*)wv_b, bv, vf);

    // 3) RoPE on q,k
    rope_kernel<<<TOKENS, 256, 0, stream>>>(qf, kf);

    // 4) per-token head-mixing attention -> bf16
    attn_kernel<<<TOKENS, 256, 0, stream>>>(qf, kf, vf, at_b);

    // 5) output projection (WMMA)
    gemm_bf16_wmma_kernel<<<ggrid, 256, 0, stream>>>(
        (const unsigned int*)at_b, (const unsigned int*)wo_b, bo, proj);

    // 6) residual + LayerNorm
    ln_kernel<<<TOKENS, 256, 0, stream>>>(x, proj, ln_g, ln_b, out);
}
